// Experts_21217138442511
// MI455X (gfx1250) — matrix operand order repaired and checked by reading, not compiled, not run on hardware
//
#include <hip/hip_runtime.h>
#include <hip/hip_bf16.h>
#include <cmath>

// MoE grouped MLP: y = GELU(x @ W1 + b1) @ W2 + b2, per expert.
//
// Pipeline (all on `stream`, serialized):
//   1) cvt_x:        x fp32 -> xb bf16                     (ws, 128 MB)
//   2) transpose_cvt: W1 -> W1T [F][D] bf16, W2 -> W2T [D][F] bf16 (ws, 64+64 MB)
//   3) per expert e: gemm1 (bf16 WMMA, TDM-staged tiles, fused bias+erf-GELU)
//                    -> h bf16 (ws, 64 MB, reused per expert)
//                    gemm2 (bf16 WMMA, TDM-staged tiles, fused bias) -> y fp32
//
// GEMM tiles are staged into LDS by the Tensor Data Mover (tensor_load_to_lds,
// TENSORcnt / s_wait_tensorcnt), double-buffered, so the k-loop contains only
// TDM issues, ds_load_b128 fragment reads and v_wmma_f32_16x16x32_bf16.

typedef __bf16 bf16_t;
typedef __attribute__((ext_vector_type(16))) __bf16 v16bf;
typedef __attribute__((ext_vector_type(8)))  __bf16 v8bf;
typedef __attribute__((ext_vector_type(4)))  __bf16 v4bf;
typedef __attribute__((ext_vector_type(8)))  float  v8f;
typedef __attribute__((ext_vector_type(4)))  unsigned int v4u;
typedef __attribute__((ext_vector_type(8)))  int v8i;
typedef __attribute__((ext_vector_type(4)))  int v4i;

constexpr int Bc = 4, Ec = 8, Tc = 2048, Dc = 1024, Fc = 4096;
constexpr int BM = 128, BN = 128, BK = 64;   // workgroup tile / K chunk
constexpr int LDA = BK + 8;                  // padded LDS strides (bank spread)
constexpr int LDB = BK + 8;

// ---------------------------------------------------------------- TDM helper
// Copy a 128-row x 64-col bf16 tile (row stride `stride_elems` in memory) to
// LDS at byte offset `lds_addr`, inserting 16 pad bytes after every 128-byte
// row so the LDS row stride is LDA*2 = 144 bytes.
// D# layout per CDNA5 ISA.
#if __has_include(<hip/amd_detail/amd_gfx1250_TDM.h>)
#define TDM_CALL(G0, G1, Z4)                                            \
  do { v8i _z8 = {0, 0, 0, 0, 0, 0, 0, 0};                              \
       __builtin_amdgcn_tensor_load_to_lds(G0, G1, Z4, Z4, _z8, 0); } while (0)
#else
#define TDM_CALL(G0, G1, Z4)                                            \
  __builtin_amdgcn_tensor_load_to_lds(G0, G1, Z4, Z4, 0)
#endif

__device__ __forceinline__ void tdm_load_tile_128x64(
    unsigned lds_addr, const bf16_t* gptr, unsigned stride_elems) {
  const unsigned long long ga = (unsigned long long)(const void*)gptr;
  // group0: count=1 | lds_addr | global_addr[56:0] | type=2 (bits 127:126)
  v4u g0 = { 1u, lds_addr, (unsigned)ga, (unsigned)(ga >> 32) | (2u << 30) };
  // group1:
  //   w0: data_size=1(2B)<<16 | pad_en<<20 | pad_interval=4(32 dw)<<22
  //       | pad_amount=3(4 dw)<<25 ; workgroup_mask=0
  //   tensor_dim0 = stride_elems, tensor_dim1 = 0x40000000 (no OOB),
  //   tile_dim0 = 64 (K, contiguous), tile_dim1 = 128 (rows), tile_dim2 = 0,
  //   tensor_dim0_stride = stride_elems, dim1_stride = 0.
  const unsigned w0 = (1u << 16) | (1u << 20) | (4u << 22) | (3u << 25);
  v8i g1 = { (int)w0,
             (int)((stride_elems & 0xFFFFu) << 16),        // barAddr=0 | td0.lo
             (int)(stride_elems >> 16),                    // td0.hi | td1.lo(=0)
             (int)((0x40000000u >> 16) | (64u << 16)),     // td1.hi | tile_dim0
             (int)BM,                                      // tile_dim1 | tile_dim2=0
             (int)stride_elems,                            // td0_stride lo
             0, 0 };
  v4i z4 = {0, 0, 0, 0};
  TDM_CALL(g0, g1, z4);
}

__device__ __forceinline__ unsigned lds_off(const void* p) {
  return (unsigned)(unsigned long long)p;   // flat shared addr: low 32 = LDS offset
}

// ------------------------------------------------------------ WMMA fragments
__device__ __forceinline__ v8f zero_v8f() {
  v8f z;
#pragma unroll
  for (int i = 0; i < 8; ++i) z[i] = 0.0f;
  return z;
}

__device__ __forceinline__ v16bf cat16(v8bf lo, v8bf hi) {
  return __builtin_shufflevector(lo, hi, 0, 1, 2, 3, 4, 5, 6, 7,
                                          8, 9, 10, 11, 12, 13, 14, 15);
}

// A fragment (16x32 bf16): lanes 0-15 hold K k0+0..7 / k0+16..23 of row M=lane,
// lanes 16-31 the +8 halves. As row-major [row][k], stride LDA.
__device__ __forceinline__ v16bf load_a_frag(const bf16_t* As, int row, int k0, int lane) {
  const bf16_t* p = As + row * LDA + k0 + ((lane & 16) ? 8 : 0);
  return cat16(*(const v8bf*)(p), *(const v8bf*)(p + 16));
}

// B fragment (32x16 bf16): lane%16 = column N; lanes 0-15 K k0+0..7 / k0+16..23,
// lanes 16-31 the +8 halves. Bs stored transposed [n][k], stride LDB.
__device__ __forceinline__ v16bf load_b_frag(const bf16_t* Bs, int col, int k0, int lane) {
  const bf16_t* p = Bs + col * LDB + k0 + ((lane & 16) ? 8 : 0);
  return cat16(*(const v8bf*)(p), *(const v8bf*)(p + 16));
}

// ------------------------------------------------------------- prep kernels
__global__ __launch_bounds__(256) void cvt_x_bf16(
    const float* __restrict__ src, bf16_t* __restrict__ dst) {
  const long i = ((long)blockIdx.x * 256 + threadIdx.x) * 4;
  const float4 v = *(const float4*)(src + i);
  v4bf o = { (bf16_t)v.x, (bf16_t)v.y, (bf16_t)v.z, (bf16_t)v.w };
  *(v4bf*)(dst + i) = o;
}

// [R][C] fp32 (per expert z) -> [C][R] bf16.
__global__ __launch_bounds__(256) void transpose_cvt(
    const float* __restrict__ src, bf16_t* __restrict__ dst, int R, int C) {
  __shared__ bf16_t tile[32][33];
  const long base = (long)blockIdx.z * R * C;
  const int r0 = blockIdx.y * 32, c0 = blockIdx.x * 32;
  const int tx = threadIdx.x & 31, ty = threadIdx.x >> 5;   // 32 x 8
#pragma unroll
  for (int i = 0; i < 32; i += 8)
    tile[ty + i][tx] = (bf16_t)src[base + (long)(r0 + ty + i) * C + c0 + tx];
  __syncthreads();
#pragma unroll
  for (int i = 0; i < 32; i += 8)
    dst[base + (long)(c0 + ty + i) * R + r0 + tx] = tile[tx][ty + i];
}

// ---------------- GEMM1: h = gelu(xb @ W1T^T + b1), h bf16 ------------------
__global__ __launch_bounds__(256) void moe_gemm1(
    const bf16_t* __restrict__ xb, const bf16_t* __restrict__ w1t,
    const float* __restrict__ b1, bf16_t* __restrict__ h, int e)
{
  __shared__ bf16_t As[2][BM * LDA];
  __shared__ bf16_t Bs[2][BN * LDB];

  const int rb = blockIdx.y, cb = blockIdx.x;
  const int tid = threadIdx.x;
  const int lane = tid & 31;
  const int wave = tid >> 5;
  const int wm = (wave >> 2) * 64;
  const int wn = (wave & 3) * 32;

  const int g0 = rb * BM;                      // expert-local row base
  const int bidx = g0 >> 11;
  const int t0 = g0 & (Tc - 1);
  const long memrow0 = (long)(bidx * Ec + e) * Tc + t0;
  const bf16_t* abase = xb + memrow0 * Dc;
  const int colbase = cb * BN;
  const bf16_t* bbase = w1t + ((long)e * Fc + colbase) * Dc;

  v8f acc[4][2];
#pragma unroll
  for (int mt = 0; mt < 4; ++mt)
#pragma unroll
    for (int nt = 0; nt < 2; ++nt) acc[mt][nt] = zero_v8f();

  constexpr int NK = Dc / BK;                  // 16
  if (tid == 0) {
    tdm_load_tile_128x64(lds_off(&As[0][0]), abase, Dc);
    tdm_load_tile_128x64(lds_off(&Bs[0][0]), bbase, Dc);
  }
  for (int it = 0; it < NK; ++it) {
    const int buf = it & 1;
    if (tid == 0) {
      if (it + 1 < NK) {
        tdm_load_tile_128x64(lds_off(&As[buf ^ 1][0]), abase + (it + 1) * BK, Dc);
        tdm_load_tile_128x64(lds_off(&Bs[buf ^ 1][0]), bbase + (it + 1) * BK, Dc);
        __builtin_amdgcn_s_wait_tensorcnt(2);  // current pair done, next in flight
      } else {
        __builtin_amdgcn_s_wait_tensorcnt(0);
      }
    }
    __syncthreads();
#pragma unroll
    for (int ks = 0; ks < BK; ks += 32) {
      v16bf afr[4], bfr[2];
#pragma unroll
      for (int mt = 0; mt < 4; ++mt)
        afr[mt] = load_a_frag(As[buf], wm + mt * 16 + (lane & 15), ks, lane);
#pragma unroll
      for (int nt = 0; nt < 2; ++nt)
        bfr[nt] = load_b_frag(Bs[buf], wn + nt * 16 + (lane & 15), ks, lane);
#pragma unroll
      for (int mt = 0; mt < 4; ++mt)
#pragma unroll
        for (int nt = 0; nt < 2; ++nt)
          acc[mt][nt] = __builtin_amdgcn_wmma_f32_16x16x32_bf16(
              false, afr[mt], false, bfr[nt], (short)0, acc[mt][nt], false, false);
    }
    __syncthreads();                           // buffer reuse fence
  }

  const float* b1e = b1 + (long)e * Fc;
#pragma unroll
  for (int nt = 0; nt < 2; ++nt) {
    const int f = colbase + wn + nt * 16 + (lane & 15);
    const float bias = b1e[f];
#pragma unroll
    for (int mt = 0; mt < 4; ++mt) {
#pragma unroll
      for (int r = 0; r < 8; ++r) {
        const int m = wm + mt * 16 + r + ((lane & 16) ? 8 : 0);
        float v = acc[mt][nt][r] + bias;
        v = 0.5f * v * (1.0f + erff(v * 0.70710678118654752f));
        h[(long)(g0 + m) * Fc + f] = (bf16_t)v;
      }
    }
  }
}

// ---------------- GEMM2: y = h @ W2T^T + b2 (fp32 out) ----------------------
__global__ __launch_bounds__(256) void moe_gemm2(
    const bf16_t* __restrict__ h, const bf16_t* __restrict__ w2t,
    const float* __restrict__ b2, float* __restrict__ y, int e)
{
  __shared__ bf16_t As[2][BM * LDA];
  __shared__ bf16_t Bs[2][BN * LDB];

  const int rb = blockIdx.y, cb = blockIdx.x;
  const int tid = threadIdx.x;
  const int lane = tid & 31;
  const int wave = tid >> 5;
  const int wm = (wave >> 2) * 64;
  const int wn = (wave & 3) * 32;

  const int g0 = rb * BM;
  const int bidx = g0 >> 11;
  const int t0 = g0 & (Tc - 1);
  const long memrow0 = (long)(bidx * Ec + e) * Tc + t0;
  const bf16_t* abase = h + (long)g0 * Fc;
  const int colbase = cb * BN;
  const bf16_t* bbase = w2t + ((long)e * Dc + colbase) * Fc;

  v8f acc[4][2];
#pragma unroll
  for (int mt = 0; mt < 4; ++mt)
#pragma unroll
    for (int nt = 0; nt < 2; ++nt) acc[mt][nt] = zero_v8f();

  constexpr int NK = Fc / BK;                  // 64
  if (tid == 0) {
    tdm_load_tile_128x64(lds_off(&As[0][0]), abase, Fc);
    tdm_load_tile_128x64(lds_off(&Bs[0][0]), bbase, Fc);
  }
  for (int it = 0; it < NK; ++it) {
    const int buf = it & 1;
    if (tid == 0) {
      if (it + 1 < NK) {
        tdm_load_tile_128x64(lds_off(&As[buf ^ 1][0]), abase + (it + 1) * BK, Fc);
        tdm_load_tile_128x64(lds_off(&Bs[buf ^ 1][0]), bbase + (it + 1) * BK, Fc);
        __builtin_amdgcn_s_wait_tensorcnt(2);
      } else {
        __builtin_amdgcn_s_wait_tensorcnt(0);
      }
    }
    __syncthreads();
#pragma unroll
    for (int ks = 0; ks < BK; ks += 32) {
      v16bf afr[4], bfr[2];
#pragma unroll
      for (int mt = 0; mt < 4; ++mt)
        afr[mt] = load_a_frag(As[buf], wm + mt * 16 + (lane & 15), ks, lane);
#pragma unroll
      for (int nt = 0; nt < 2; ++nt)
        bfr[nt] = load_b_frag(Bs[buf], wn + nt * 16 + (lane & 15), ks, lane);
#pragma unroll
      for (int mt = 0; mt < 4; ++mt)
#pragma unroll
        for (int nt = 0; nt < 2; ++nt)
          acc[mt][nt] = __builtin_amdgcn_wmma_f32_16x16x32_bf16(
              false, afr[mt], false, bfr[nt], (short)0, acc[mt][nt], false, false);
    }
    __syncthreads();
  }

  const float* b2e = b2 + (long)e * Dc;
#pragma unroll
  for (int nt = 0; nt < 2; ++nt) {
    const int d = colbase + wn + nt * 16 + (lane & 15);
    const float bias = b2e[d];
#pragma unroll
    for (int mt = 0; mt < 4; ++mt) {
#pragma unroll
      for (int r = 0; r < 8; ++r) {
        const int m = wm + mt * 16 + r + ((lane & 16) ? 8 : 0);
        y[(memrow0 + m) * (long)Dc + d] = acc[mt][nt][r] + bias;
      }
    }
  }
}

// --------------------------------------------------------------------- host
extern "C" void kernel_launch(void* const* d_in, const int* in_sizes, int n_in,
                              void* d_out, int out_size, void* d_ws, size_t ws_size,
                              hipStream_t stream) {
  const float* x  = (const float*)d_in[0];
  const float* W1 = (const float*)d_in[1];
  const float* b1 = (const float*)d_in[2];
  const float* W2 = (const float*)d_in[3];
  const float* b2 = (const float*)d_in[4];
  float* y = (float*)d_out;

  // Workspace layout (320 MB total):
  bf16_t* xb  = (bf16_t*)d_ws;                         // 128 MB
  bf16_t* w1t = xb  + (long)Bc * Ec * Tc * Dc;          //  64 MB  [E][F][D]
  bf16_t* w2t = w1t + (long)Ec * Fc * Dc;               //  64 MB  [E][D][F]
  bf16_t* h   = w2t + (long)Ec * Dc * Fc;               //  64 MB  per-expert

  const dim3 blk(256, 1, 1);
  const long nx = (long)Bc * Ec * Tc * Dc;
  cvt_x_bf16<<<dim3((unsigned)(nx / 1024)), blk, 0, stream>>>(x, xb);
  transpose_cvt<<<dim3(Fc / 32, Dc / 32, Ec), blk, 0, stream>>>(W1, w1t, Dc, Fc);
  transpose_cvt<<<dim3(Dc / 32, Fc / 32, Ec), blk, 0, stream>>>(W2, w2t, Fc, Dc);

  const dim3 g1(Fc / BN, (Bc * Tc) / BM, 1);   // 32 x 64
  const dim3 g2(Dc / BN, (Bc * Tc) / BM, 1);   //  8 x 64
  for (int e = 0; e < Ec; ++e) {
    moe_gemm1<<<g1, blk, 0, stream>>>(xb, w1t, b1, h, e);
    moe_gemm2<<<g2, blk, 0, stream>>>(h, w2t, b2, y, e);
  }
}
